// _SelfAttSqrtn_50302656971167
// MI455X (gfx1250) — matrix operand order repaired and checked by reading, not compiled, not run on hardware
//
#include <hip/hip_runtime.h>

// Problem constants (must match reference)
#define B_      4
#define L_      4096
#define DMODEL  1024
#define H_      16
#define SEG_    64
#define HD_     64
#define NSEG_   64
#define NEGV    (-1.0e10f)
#define NINF    (-3.0e38f)

typedef __attribute__((ext_vector_type(16))) __bf16 v16bf;
typedef __attribute__((ext_vector_type(8)))  float  v8f;

union FragU { v16bf v; unsigned short u[16]; };

__device__ __forceinline__ unsigned short f2bf(float f) {
  unsigned int u = __float_as_uint(f);
  u += 0x7fffu + ((u >> 16) & 1u);        // round-to-nearest-even
  return (unsigned short)(u >> 16);
}
__device__ __forceinline__ float bf2f(unsigned short h) {
  return __uint_as_float(((unsigned int)h) << 16);
}

__device__ __forceinline__ v8f wmma_bf16(v16bf a, v16bf b, v8f c) {
  return __builtin_amdgcn_wmma_f32_16x16x32_bf16(
      /*neg_a=*/false, a, /*neg_b=*/false, b,
      /*c_mod=*/(short)0, c, /*reuse_a=*/false, /*reuse_b=*/false);
}

// ---- CDNA5 async global->LDS staging (ASYNCcnt path), with safe fallback ----
#if defined(__AMDGCN__) && __has_builtin(__builtin_amdgcn_global_load_async_to_lds_b128)
#define USE_ASYNC_LDS 1
#else
#define USE_ASYNC_LDS 0
#endif

// Pointee type per hipcc diagnostic: '__attribute__((__vector_size__(4*sizeof(int)))) int'
typedef int v4i_async __attribute__((vector_size(16)));
typedef __attribute__((address_space(1))) v4i_async* gasync_ptr;
typedef __attribute__((address_space(3))) v4i_async* lasync_ptr;

__device__ __forceinline__ void async_copy16(const void* gsrc, void* lds_dst) {
#if USE_ASYNC_LDS
  __builtin_amdgcn_global_load_async_to_lds_b128(
      (gasync_ptr)(gsrc), (lasync_ptr)(lds_dst), /*offset=*/0, /*cpol=*/0);
#else
  *(uint4*)lds_dst = *(const uint4*)gsrc;
#endif
}

__device__ __forceinline__ void async_wait0() {
#if USE_ASYNC_LDS
#if __has_builtin(__builtin_amdgcn_s_wait_asynccnt)
  __builtin_amdgcn_s_wait_asynccnt(0);
#else
  asm volatile("s_wait_asynccnt 0x0" ::: "memory");
#endif
#endif
}

// ---- Fragment loaders (contiguous k order: k = 16*(lane/16) + e) ----------------------------
// A matrix 16x32 bf16: lanes 0-15 -> M=lane, K=0..15; lanes 16-31 -> M=lane-16, K=16..31
__device__ __forceinline__ v16bf load_frag_a(const unsigned short* tile, int ldk) {
  int lane = threadIdx.x & 31;
  int r = lane & 15, hi = lane >> 4;
  FragU f;
#pragma unroll
  for (int e = 0; e < 16; ++e) {
    int k = (hi << 4) + e;
    f.u[e] = tile[r * ldk + k];
  }
  return f.v;
}
// B matrix 32x16 bf16 from Bt[n][k] row-major: lanes 0-15 -> N=lane, K=0..15; lanes 16-31 -> N=lane-16, K=16..31
__device__ __forceinline__ v16bf load_frag_b(const unsigned short* tile, int ldk) {
  int lane = threadIdx.x & 31;
  int n = lane & 15, hi = lane >> 4;
  FragU f;
#pragma unroll
  for (int e = 0; e < 16; ++e) {
    int k = (hi << 4) + e;
    f.u[e] = tile[n * ldk + k];
  }
  return f.v;
}
// B matrix 32x16 bf16 from tile stored as [k][n] row-major (natural V / span_val layout)
__device__ __forceinline__ v16bf load_frag_bT(const unsigned short* tile, int ldn, int ncol) {
  int lane = threadIdx.x & 31;
  int n = ncol + (lane & 15), hi = lane >> 4;
  FragU f;
#pragma unroll
  for (int e = 0; e < 16; ++e) {
    int k = (hi << 4) + e;
    f.u[e] = tile[k * ldn + n];
  }
  return f.v;
}

// ---- Kernel 1a: convert + transpose weights to bf16 -------------------------
// wq/wk/wv: [d][h*f] -> wqkv_t: [n(3072)][k(1024)]  (q cols | k cols | v cols)
// wo:       [h*f][dout] -> wo_t: [n(1024)][k(1024)]
__global__ __launch_bounds__(256) void convert_weights_kernel(
    const float* __restrict__ wq, const float* __restrict__ wk,
    const float* __restrict__ wv, const float* __restrict__ wo,
    unsigned short* __restrict__ wqkv_t, unsigned short* __restrict__ wo_t) {
  int i = blockIdx.x * blockDim.x + threadIdx.x;
  if (i >= DMODEL * DMODEL) return;
  int n = i / DMODEL, k = i % DMODEL;
  size_t src = (size_t)k * DMODEL + n;
  wqkv_t[(size_t)n * DMODEL + k]                = f2bf(wq[src]);
  wqkv_t[(size_t)(n + DMODEL) * DMODEL + k]     = f2bf(wk[src]);
  wqkv_t[(size_t)(n + 2 * DMODEL) * DMODEL + k] = f2bf(wv[src]);
  wo_t[(size_t)n * DMODEL + k]                  = f2bf(wo[src]);
}

// ---- Kernel 1b: convert activations x (f32) -> bf16, vectorized -------------
__global__ __launch_bounds__(256) void convert_x_kernel(
    const float* __restrict__ x, unsigned short* __restrict__ xb, int n8) {
  int i = blockIdx.x * blockDim.x + threadIdx.x;  // one thread = 8 elements
  if (i >= n8) return;
  const float4* xf = (const float4*)x;
  float4 a = xf[2 * i], b = xf[2 * i + 1];
  union { unsigned short u[8]; uint4 v; } o;
  o.u[0] = f2bf(a.x); o.u[1] = f2bf(a.y); o.u[2] = f2bf(a.z); o.u[3] = f2bf(a.w);
  o.u[4] = f2bf(b.x); o.u[5] = f2bf(b.y); o.u[6] = f2bf(b.z); o.u[7] = f2bf(b.w);
  ((uint4*)xb)[i] = o.v;
}

// ---- Kernel 2/5: tiled WMMA GEMM  C[M,N] = A[M,K] * Bt[N,K]^T ---------------
// Block tile 128x128, K-step 32, 256 threads = 8 waves (4 M-groups x 2 N-groups).
// Double-buffered: async engine fills buffer (cur^1) while WMMAs consume cur.
template <bool OUT_BF16>
__global__ __launch_bounds__(256) void gemm_wmma_kernel(
    const unsigned short* __restrict__ A, const unsigned short* __restrict__ Bt,
    void* __restrict__ Cv, int M, int N, int K, float scale, int scale_nlimit) {
  __shared__ __align__(16) unsigned short As[2][128 * 32];
  __shared__ __align__(16) unsigned short Bs[2][128 * 32];
  const int m0 = blockIdx.x * 128, n0 = blockIdx.y * 128;
  const int t = threadIdx.x, w = t >> 5, lane = t & 31;
  const int mw = (w & 3) * 32, nw = (w >> 2) * 64;

  v8f acc[2][4];
#pragma unroll
  for (int i = 0; i < 2; ++i)
#pragma unroll
    for (int j = 0; j < 4; ++j) acc[i][j] = (v8f){};

  auto stage = [&](int buf, int k0) {
#pragma unroll
    for (int c = t; c < 512; c += 256) {
      int row = c >> 2, kp = (c & 3) << 3;
      async_copy16(&A[(size_t)(m0 + row) * K + k0 + kp],  &As[buf][row * 32 + kp]);
      async_copy16(&Bt[(size_t)(n0 + row) * K + k0 + kp], &Bs[buf][row * 32 + kp]);
    }
  };

  stage(0, 0);
  async_wait0();
  __syncthreads();

  int cur = 0;
  for (int k0 = 0; k0 < K; k0 += 32, cur ^= 1) {
    if (k0 + 32 < K) stage(cur ^ 1, k0 + 32);  // prefetch next tile (overlaps WMMAs)

    v16bf a0 = load_frag_a(&As[cur][(mw + 0) * 32], 32);
    v16bf a1 = load_frag_a(&As[cur][(mw + 16) * 32], 32);
#pragma unroll
    for (int j = 0; j < 4; ++j) {
      v16bf b = load_frag_b(&Bs[cur][(nw + j * 16) * 32], 32);
      acc[0][j] = wmma_bf16(a0, b, acc[0][j]);
      acc[1][j] = wmma_bf16(a1, b, acc[1][j]);
    }

    if (k0 + 32 < K) {
      async_wait0();
      __syncthreads();
    }
  }

  const int nloc = lane & 15, hiL = lane >> 4;
#pragma unroll
  for (int i = 0; i < 2; ++i)
#pragma unroll
    for (int j = 0; j < 4; ++j)
#pragma unroll
      for (int r = 0; r < 8; ++r) {
        int row = m0 + mw + i * 16 + r + hiL * 8;
        int col = n0 + nw + j * 16 + nloc;
        float v = acc[i][j][r];
        if (col < scale_nlimit) v *= scale;
        if (OUT_BF16) ((unsigned short*)Cv)[(size_t)row * N + col] = f2bf(v);
        else          ((float*)Cv)[(size_t)row * N + col] = v;
      }
}

// ---- Kernel 3: span summaries (mq, span_key, span_val) ----------------------
// One block of 64 threads per (b, seg, h). Tiny FLOPs -> VALU is fine.
__global__ __launch_bounds__(64) void span_kernel(
    const unsigned short* __restrict__ qkv,
    unsigned short* __restrict__ span_key, unsigned short* __restrict__ span_val) {
  int bid = blockIdx.x;
  int h = bid % H_; int bs = bid / H_;
  int s = bs % NSEG_; int b = bs / NSEG_;
  int t = threadIdx.x;  // 0..63 (column index)
  const size_t base = ((size_t)b * L_ + (size_t)s * SEG_) * 3072 + (size_t)h * HD_;

  __shared__ float mq[HD_];
  __shared__ float lw[SEG_];

  float mmax = NINF, kmax = NINF;
  for (int r = 0; r < SEG_; ++r) {
    mmax = fmaxf(mmax, bf2f(qkv[base + (size_t)r * 3072 + t]));
    kmax = fmaxf(kmax, bf2f(qkv[base + (size_t)r * 3072 + 1024 + t]));
  }
  mq[t] = mmax;
  __syncthreads();

  float l = 0.f;  // logit for key row t: mq . ck[t]
  for (int d = 0; d < HD_; ++d)
    l += mq[d] * bf2f(qkv[base + (size_t)t * 3072 + 1024 + d]);
  lw[t] = l;
  __syncthreads();

  float mx = NINF;
  for (int d = 0; d < SEG_; ++d) mx = fmaxf(mx, lw[d]);
  float sum = 0.f;
  for (int d = 0; d < SEG_; ++d) sum += __expf(lw[d] - mx);
  float wgt = __expf(l - mx) / sum;
  __syncthreads();
  lw[t] = wgt;
  __syncthreads();

  float sv = 0.f;  // span_val column t
  for (int r = 0; r < SEG_; ++r)
    sv += lw[r] * bf2f(qkv[base + (size_t)r * 3072 + 2048 + t]);

  size_t o = (((size_t)b * NSEG_ + s) * H_ + h) * HD_ + t;
  span_val[o] = f2bf(sv);
  span_key[o] = f2bf(kmax);
}

// ---- Kernel 4: segment attention (local + prev spans), fused softmax --------
// One block of 128 threads (4 waves) per (b, seg, h). Wave w owns query rows [16w,16w+16).
__global__ __launch_bounds__(128) void attn_kernel(
    const unsigned short* __restrict__ qkv,
    const unsigned short* __restrict__ span_key,
    const unsigned short* __restrict__ span_val,
    unsigned short* __restrict__ ctx) {
  int bid = blockIdx.x;
  int h = bid % H_; int bs = bid / H_;
  int s = bs % NSEG_; int b = bs / NSEG_;

  __shared__ __align__(16) unsigned short Qs[SEG_ * HD_];
  __shared__ __align__(16) unsigned short Ks[SEG_ * HD_];
  __shared__ __align__(16) unsigned short Vs[SEG_ * HD_];
  __shared__ __align__(16) unsigned short SKs[NSEG_ * HD_];
  __shared__ __align__(16) unsigned short SVs[NSEG_ * HD_];
  __shared__ __align__(16) unsigned short AttB[SEG_ * 128];  // [row][0..63 local | 64..127 prev]

  const int t = threadIdx.x, w = t >> 5, lane = t & 31;
  const size_t base = ((size_t)b * L_ + (size_t)s * SEG_) * 3072 + (size_t)h * HD_;

#pragma unroll
  for (int c = t; c < 512; c += 128) {
    int row = c >> 3, dp = (c & 7) << 3;
    async_copy16(&qkv[base + (size_t)row * 3072 + dp],        &Qs[row * 64 + dp]);
    async_copy16(&qkv[base + (size_t)row * 3072 + 1024 + dp], &Ks[row * 64 + dp]);
    async_copy16(&qkv[base + (size_t)row * 3072 + 2048 + dp], &Vs[row * 64 + dp]);
    size_t sb = (((size_t)b * NSEG_ + row) * H_ + h) * HD_ + dp;
    async_copy16(&span_key[sb], &SKs[row * 64 + dp]);
    async_copy16(&span_val[sb], &SVs[row * 64 + dp]);
  }
  async_wait0();
  __syncthreads();

  const int mrow = w * 16;
  v16bf aq0 = load_frag_a(&Qs[mrow * 64 + 0], 64);
  v16bf aq1 = load_frag_a(&Qs[mrow * 64 + 32], 64);

  v8f Lc[4], Pc[4];
#pragma unroll
  for (int j = 0; j < 4; ++j) {
    v8f c = (v8f){};
    c = wmma_bf16(aq0, load_frag_b(&Ks[(j * 16) * 64 + 0], 64), c);
    c = wmma_bf16(aq1, load_frag_b(&Ks[(j * 16) * 64 + 32], 64), c);
    Lc[j] = c;
    c = (v8f){};
    c = wmma_bf16(aq0, load_frag_b(&SKs[(j * 16) * 64 + 0], 64), c);
    c = wmma_bf16(aq1, load_frag_b(&SKs[(j * 16) * 64 + 32], 64), c);
    Pc[j] = c;
  }

  const int nloc = lane & 15, hiL = lane >> 4;
  // Masks: local causal (key > q) ; prev allowed only for span t < s.
#pragma unroll
  for (int j = 0; j < 4; ++j)
#pragma unroll
    for (int r = 0; r < 8; ++r) {
      int rowl = mrow + r + hiL * 8;
      int keyl = j * 16 + nloc;
      if (keyl > rowl) Lc[j][r] += NEGV;
      if (keyl >= s)   Pc[j][r] += NEGV;
    }

  // Row softmax over 128 joint logits: register combine across 8 tiles,
  // then shfl-xor reduce across the 16-lane N group.
#pragma unroll
  for (int r = 0; r < 8; ++r) {
    float mx = NINF;
#pragma unroll
    for (int j = 0; j < 4; ++j) mx = fmaxf(mx, fmaxf(Lc[j][r], Pc[j][r]));
    for (int off = 8; off; off >>= 1) mx = fmaxf(mx, __shfl_xor(mx, off, 32));
    float sum = 0.f;
#pragma unroll
    for (int j = 0; j < 4; ++j) {
      float el = __expf(Lc[j][r] - mx);
      float ep = __expf(Pc[j][r] - mx);
      Lc[j][r] = el; Pc[j][r] = ep;
      sum += el + ep;
    }
    for (int off = 8; off; off >>= 1) sum += __shfl_xor(sum, off, 32);
    float inv = 1.0f / sum;
#pragma unroll
    for (int j = 0; j < 4; ++j) { Lc[j][r] *= inv; Pc[j][r] *= inv; }
  }

  // Re-fragment attention weights: C layout -> LDS (bf16) -> A layout.
#pragma unroll
  for (int j = 0; j < 4; ++j)
#pragma unroll
    for (int r = 0; r < 8; ++r) {
      int rowl = mrow + r + hiL * 8;
      AttB[rowl * 128 + j * 16 + nloc]      = f2bf(Lc[j][r]);
      AttB[rowl * 128 + 64 + j * 16 + nloc] = f2bf(Pc[j][r]);
    }
  __syncthreads();

  v16bf al0 = load_frag_a(&AttB[mrow * 128 + 0],  128);
  v16bf al1 = load_frag_a(&AttB[mrow * 128 + 32], 128);
  v16bf ap0 = load_frag_a(&AttB[mrow * 128 + 64], 128);
  v16bf ap1 = load_frag_a(&AttB[mrow * 128 + 96], 128);

#pragma unroll
  for (int j = 0; j < 4; ++j) {
    v8f O = (v8f){};
    O = wmma_bf16(al0, load_frag_bT(&Vs[0],        64, j * 16), O);
    O = wmma_bf16(al1, load_frag_bT(&Vs[32 * 64],  64, j * 16), O);
    O = wmma_bf16(ap0, load_frag_bT(&SVs[0],       64, j * 16), O);
    O = wmma_bf16(ap1, load_frag_bT(&SVs[32 * 64], 64, j * 16), O);
#pragma unroll
    for (int r = 0; r < 8; ++r) {
      int rowl = mrow + r + hiL * 8;
      size_t o = ((size_t)b * L_ + (size_t)s * SEG_ + rowl) * DMODEL
               + (size_t)h * HD_ + j * 16 + nloc;
      ctx[o] = f2bf(O[r]);
    }
  }
}

// ---- Host-side orchestration ------------------------------------------------
extern "C" void kernel_launch(void* const* d_in, const int* in_sizes, int n_in,
                              void* d_out, int out_size, void* d_ws, size_t ws_size,
                              hipStream_t stream) {
  const float* x  = (const float*)d_in[0];
  const float* wq = (const float*)d_in[1];
  const float* wk = (const float*)d_in[2];
  const float* wv = (const float*)d_in[3];
  const float* wo = (const float*)d_in[4];
  float* out = (float*)d_out;

  // Workspace carve-out (bf16 = unsigned short), 256B aligned slices.
  size_t off = 0;
  auto carve = [&](size_t bytes) -> char* {
    char* p = (char*)d_ws + off;
    off += (bytes + 255) & ~(size_t)255;
    return p;
  };
  unsigned short* wqkv_t = (unsigned short*)carve((size_t)3 * DMODEL * DMODEL * 2);
  unsigned short* wo_t   = (unsigned short*)carve((size_t)DMODEL * DMODEL * 2);
  unsigned short* xb     = (unsigned short*)carve((size_t)B_ * L_ * DMODEL * 2);
  unsigned short* qkv    = (unsigned short*)carve((size_t)B_ * L_ * 3 * DMODEL * 2);
  unsigned short* skey   = (unsigned short*)carve((size_t)B_ * NSEG_ * H_ * HD_ * 2);
  unsigned short* sval   = (unsigned short*)carve((size_t)B_ * NSEG_ * H_ * HD_ * 2);
  unsigned short* ctx    = (unsigned short*)carve((size_t)B_ * L_ * DMODEL * 2);

  const int M = B_ * L_;             // 16384
  const float qscale = 0.125f;       // 1/sqrt(HD)
  const int n_x8 = (B_ * L_ * DMODEL) / 8;

  // 1) Weights -> bf16 (transposed for Bt[N][K]); activations -> bf16.
  convert_weights_kernel<<<(DMODEL * DMODEL + 255) / 256, 256, 0, stream>>>(
      wq, wk, wv, wo, wqkv_t, wo_t);
  convert_x_kernel<<<(n_x8 + 255) / 256, 256, 0, stream>>>(x, xb, n_x8);

  // 2) Fused QKV projection: qkv[M, 3072] = xb[M,1024] * wqkv_t^T (q cols scaled).
  gemm_wmma_kernel<true><<<dim3(M / 128, (3 * DMODEL) / 128), 256, 0, stream>>>(
      xb, wqkv_t, (void*)qkv, M, 3 * DMODEL, DMODEL, qscale, DMODEL);

  // 3) Span summaries per (b, seg, h).
  span_kernel<<<B_ * NSEG_ * H_, 64, 0, stream>>>(qkv, skey, sval);

  // 4) Segment attention (local causal + previous-span), fused joint softmax.
  attn_kernel<<<B_ * NSEG_ * H_, 128, 0, stream>>>(qkv, skey, sval, ctx);

  // 5) Output projection: out[M,1024] = ctx[M,1024] * wo_t^T (f32 store).
  gemm_wmma_kernel<false><<<dim3(M / 128, DMODEL / 128), 256, 0, stream>>>(
      ctx, wo_t, (void*)out, M, DMODEL, DMODEL, 1.0f, 0);
}
